// SpatialAttention_9199819948442
// MI455X (gfx1250) — matrix operand order repaired and checked by reading, not compiled, not run on hardware
//
#include <hip/hip_runtime.h>
#include <hip/hip_bf16.h>
#include <math.h>

typedef __attribute__((ext_vector_type(16))) __bf16 v16bf;
typedef __attribute__((ext_vector_type(8)))  float  v8f;
typedef __attribute__((ext_vector_type(4)))  unsigned int v4u;

static constexpr int kHW   = 4096;   // 64*64 spatial positions
static constexpr int kC    = 512;    // channels
static constexpr int kB    = 8;      // batch
static constexpr int kCPG  = 16;     // channels per group (512/32)

union FragU { v16bf f; v4u q[2]; };
union Pack8 { __bf16 h[8]; v4u q; };

// A-fragment (16x32 bf16, row-major MxK source): lane holds row m=lane&15,
// K = kb..kb+15 (elems 0..15), kb=(lane>>4)*16.
__device__ __forceinline__ v16bf load_fragA(const __bf16* p) {
  FragU u;
  u.q[0] = *(const v4u*)(p);
  u.q[1] = *(const v4u*)(p + 8);
  return u.f;
}
// B-fragment (32x16 bf16) from B^T row-major (NxK): lane holds col n=lane&15,
// K = (lane>>4)*16 + e, e=0..15 -> 32 contiguous bytes.
__device__ __forceinline__ v16bf load_fragB(const __bf16* p) {
  FragU u;
  u.q[0] = *(const v4u*)(p);
  u.q[1] = *(const v4u*)(p + 8);
  return u.f;
}
__device__ __forceinline__ v8f wmma_bf16(v16bf a, v16bf b, v8f c) {
  return __builtin_amdgcn_wmma_f32_16x16x32_bf16(false, a, false, b,
                                                 (short)0, c, false, false);
}
__device__ __forceinline__ void zero_acc(v8f& a) {
  #pragma unroll
  for (int e = 0; e < 8; e++) a[e] = 0.f;
}

// ---------------------------------------------------------------- f32 -> bf16
__global__ void cvt_f32_bf16(const float* __restrict__ src,
                             __bf16* __restrict__ dst, int n) {
  int i = blockIdx.x * blockDim.x + threadIdx.x;
  if (i < n) dst[i] = (__bf16)src[i];
}

// ------------------------------------------------------------------ GroupNorm
// One block per (batch, group): reduce 16 ch x 4096 px, write R^T (hw,c) bf16
// with two b128 stores per pixel (the group's 16 channels are contiguous).
__global__ void gn_kernel(const float* __restrict__ x,
                          const float* __restrict__ gamma,
                          const float* __restrict__ beta,
                          __bf16* __restrict__ Rt) {
  int bg = blockIdx.x;
  int b = bg >> 5, g = bg & 31;
  const float* xp = x + ((size_t)b * kC + g * kCPG) * kHW;
  int tid = threadIdx.x;
  float s = 0.f, s2 = 0.f;
  for (int i = tid; i < kCPG * kHW; i += 256) {
    float v = xp[i]; s += v; s2 += v * v;
  }
  for (int m = 1; m < 32; m <<= 1) {
    s  += __shfl_xor(s,  m, 32);
    s2 += __shfl_xor(s2, m, 32);
  }
  __shared__ float ss[8], ss2[8];
  int w = tid >> 5;
  if ((tid & 31) == 0) { ss[w] = s; ss2[w] = s2; }
  __syncthreads();
  if (tid == 0) {
    float a = 0.f, a2 = 0.f;
    for (int i = 0; i < 8; i++) { a += ss[i]; a2 += ss2[i]; }
    ss[0] = a; ss2[0] = a2;
  }
  __syncthreads();
  const float invN = 1.0f / (float)(kCPG * kHW);
  float mean = ss[0] * invN;
  float var  = ss2[0] * invN - mean * mean;
  float inv  = rsqrtf(var + 1e-6f);
  float ga[kCPG], be[kCPG];
  #pragma unroll
  for (int ch = 0; ch < kCPG; ch++) {
    ga[ch] = gamma[g * kCPG + ch] * inv;
    be[ch] = beta[g * kCPG + ch];
  }
  // per-pixel: each lane normalizes all 16 channels of its pixel, packed store
  for (int t = tid; t < kHW; t += 256) {
    Pack8 lo, hiP;
    #pragma unroll
    for (int ch = 0; ch < 8; ch++)
      lo.h[ch] = (__bf16)((xp[(size_t)ch * kHW + t] - mean) * ga[ch] + be[ch]);
    #pragma unroll
    for (int ch = 0; ch < 8; ch++)
      hiP.h[ch] = (__bf16)((xp[(size_t)(ch + 8) * kHW + t] - mean) * ga[ch + 8] + be[ch + 8]);
    __bf16* dst = Rt + ((size_t)b * kHW + t) * kC + g * kCPG;
    *(v4u*)(dst)     = lo.q;
    *(v4u*)(dst + 8) = hiP.q;
  }
}

// ------------------------------------------------------- 1x1 conv as WMMA GEMM
// C[o,t] = alpha*(W[o,:]·R[:,t] + bias[o]); B operand from R^T (hw,c).
// Block = 8 waves; wave: 16(M) x 64(N); grid (hw/64, c/128, batch).
// __launch_bounds__(256,1): lift VGPR cap so the k+1 fragment buffers stay
// in distinct registers and loads overlap the current WMMA chain.
__global__ void __launch_bounds__(256, 1)
conv1x1_gemm(const __bf16* __restrict__ Wb,
             const __bf16* __restrict__ Bt,
             const float* __restrict__ bias,
             __bf16* __restrict__ out,
             float alpha, int transposedStore) {
  int b    = blockIdx.z;
  int lane = threadIdx.x & 31;
  int w    = threadIdx.x >> 5;
  int m0   = blockIdx.y * 128 + w * 16;
  int nb   = blockIdx.x * 64;
  int lm   = lane & 15;
  int hi   = lane >> 4;
  int kbA  = hi * 16;
  int kbB  = hi * 16;
  const __bf16* Ap  = Wb + (size_t)(m0 + lm) * kC + kbA;
  const __bf16* Btb = Bt + (size_t)b * kHW * kC;
  const __bf16* Bp[4];
  #pragma unroll
  for (int i = 0; i < 4; i++)
    Bp[i] = Btb + (size_t)(nb + 16 * i + lm) * kC + kbB;
  v8f acc[4];
  #pragma unroll
  for (int i = 0; i < 4; i++) zero_acc(acc[i]);
  v16bf a = load_fragA(Ap);
  v16bf bfr[4];
  #pragma unroll
  for (int i = 0; i < 4; i++) bfr[i] = load_fragB(Bp[i]);
  #pragma unroll
  for (int kk = 0; kk < 16; kk++) {
    v16bf a_n = a;
    v16bf b_n[4];
    #pragma unroll
    for (int i = 0; i < 4; i++) b_n[i] = bfr[i];
    if (kk < 15) {
      a_n = load_fragA(Ap + (kk + 1) * 32);
      #pragma unroll
      for (int i = 0; i < 4; i++) b_n[i] = load_fragB(Bp[i] + (kk + 1) * 32);
    }
    #pragma unroll
    for (int i = 0; i < 4; i++) acc[i] = wmma_bf16(a, bfr[i], acc[i]);
    a = a_n;
    #pragma unroll
    for (int i = 0; i < 4; i++) bfr[i] = b_n[i];
  }
  #pragma unroll
  for (int i = 0; i < 4; i++) {
    int n0 = nb + 16 * i;
    if (transposedStore) {
      Pack8 pk;
      #pragma unroll
      for (int r = 0; r < 8; r++)
        pk.h[r] = (__bf16)(alpha * (acc[i][r] + bias[m0 + r + 8 * hi]));
      // rows m0+8*hi..+7 contiguous in (hw,c) layout -> one b128 store
      *(v4u*)(out + ((size_t)b * kHW + n0 + lm) * kC + m0 + 8 * hi) = pk.q;
    } else {
      #pragma unroll
      for (int r = 0; r < 8; r++) {
        int row = m0 + r + 8 * hi;
        float vv = alpha * (acc[i][r] + bias[row]);
        out[((size_t)b * kC + row) * kHW + n0 + lm] = (__bf16)vv;
      }
    }
  }
}

// --------------------------------------------------- pass A: softmax row stats
// scores[s,t] = q_s·k_t (q pre-scaled by 1/sqrt(c)). One block per (s-tile,b);
// wave w covers t in [w*512, w*512+512); online max/sumexp per row via
// 16-lane shuffles; cross-wave merge in LDS.
__global__ void __launch_bounds__(256, 1)
attn_stats(const __bf16* __restrict__ qT,
           const __bf16* __restrict__ kT,
           float* __restrict__ Ms, float* __restrict__ Ls) {
  int s0   = blockIdx.x * 16;
  int b    = blockIdx.y;
  int lane = threadIdx.x & 31;
  int w    = threadIdx.x >> 5;
  int lm   = lane & 15, hi = lane >> 4;
  int kbA  = hi * 16, kbB = hi * 16;
  const __bf16* qb = qT + (size_t)b * kHW * kC;
  const __bf16* kb = kT + (size_t)b * kHW * kC;
  v16bf afr[16];
  #pragma unroll
  for (int kk = 0; kk < 16; kk++)
    afr[kk] = load_fragA(qb + (size_t)(s0 + lm) * kC + kk * 32 + kbA);
  float mrow[8], lrow[8];
  #pragma unroll
  for (int r = 0; r < 8; r++) { mrow[r] = -3.0e38f; lrow[r] = 0.f; }
  for (int j = 0; j < 32; j++) {
    const __bf16* kp = kb + (size_t)((w * 32 + j) * 16 + lm) * kC + kbB;
    v8f acc; zero_acc(acc);
    v16bf bf = load_fragB(kp);
    #pragma unroll
    for (int kk = 0; kk < 16; kk++) {
      v16bf b_n = bf;
      if (kk < 15) b_n = load_fragB(kp + (kk + 1) * 32);
      acc = wmma_bf16(afr[kk], bf, acc);
      bf = b_n;
    }
    #pragma unroll
    for (int r = 0; r < 8; r++) {
      float v = acc[r];
      float tmax = v;
      for (int msk = 1; msk < 16; msk <<= 1)
        tmax = fmaxf(tmax, __shfl_xor(tmax, msk, 32));
      float nm = fmaxf(mrow[r], tmax);
      float e = __expf(v - nm);
      float tsum = e;
      for (int msk = 1; msk < 16; msk <<= 1)
        tsum += __shfl_xor(tsum, msk, 32);
      lrow[r] = lrow[r] * __expf(mrow[r] - nm) + tsum;
      mrow[r] = nm;
    }
  }
  __shared__ float sm[8][16], sl[8][16];
  if (lm == 0) {
    #pragma unroll
    for (int r = 0; r < 8; r++) {
      sm[w][r + 8 * hi] = mrow[r];
      sl[w][r + 8 * hi] = lrow[r];
    }
  }
  __syncthreads();
  if (threadIdx.x < 16) {
    float m = -3.0e38f;
    for (int i = 0; i < 8; i++) m = fmaxf(m, sm[i][threadIdx.x]);
    float l = 0.f;
    for (int i = 0; i < 8; i++) l += sl[i][threadIdx.x] * __expf(sm[i][threadIdx.x] - m);
    Ms[(size_t)b * kHW + s0 + threadIdx.x] = m;
    Ls[(size_t)b * kHW + s0 + threadIdx.x] = l;
  }
}

// -------------------------------------- pass B: out[c,t] = sum_s v[c,s]*P[s,t]
// One block per (t-tile of 16, batch). k-fragments for the t-tile preloaded in
// 128 VGPRs and reused across all 16 s-chunks. Each s-chunk: waves compute
// disjoint 32-row score tiles, normalize, stage P^T in LDS (b128 stores),
// then WMMA against V tiles (wave owns 64 channels). Writes out^T (hw,c) bf16.
__global__ void __launch_bounds__(256, 1)
attn_out(const __bf16* __restrict__ qT,
         const __bf16* __restrict__ kT,
         const __bf16* __restrict__ V,
         const float* __restrict__ Ms,
         const float* __restrict__ Ls,
         __bf16* __restrict__ outT) {
  int t0   = blockIdx.x * 16;
  int b    = blockIdx.y;
  int lane = threadIdx.x & 31;
  int w    = threadIdx.x >> 5;
  int lm   = lane & 15, hi = lane >> 4;
  int kbA  = hi * 16, kbB = hi * 16;
  const __bf16* qb = qT + (size_t)b * kHW * kC;
  const __bf16* kb = kT + (size_t)b * kHW * kC;
  const __bf16* vb = V  + (size_t)b * kC * kHW;
  __shared__ __attribute__((aligned(16))) __bf16 pT[16][272];  // t x s-chunk
  v16bf kfr[16];
  #pragma unroll
  for (int kk = 0; kk < 16; kk++)
    kfr[kk] = load_fragB(kb + (size_t)(t0 + lm) * kC + kk * 32 + kbB);
  v8f acc[4];
  #pragma unroll
  for (int i = 0; i < 4; i++) zero_acc(acc[i]);
  for (int chunk = 0; chunk < 16; chunk++) {
    int sbase = chunk * 256;
    if (chunk < 15) {
      // warm next chunk's q rows into cache (global_prefetch_b8)
      __builtin_prefetch(qb + (size_t)(sbase + 256 + w * 32 + lm) * kC + kbA, 0, 1);
    }
    #pragma unroll
    for (int half = 0; half < 2; half++) {
      int ss0 = sbase + w * 32 + half * 16;
      const __bf16* qp = qb + (size_t)(ss0 + lm) * kC + kbA;
      v8f sacc; zero_acc(sacc);
      v16bf a = load_fragA(qp);
      #pragma unroll
      for (int kk = 0; kk < 16; kk++) {
        v16bf a_n = a;
        if (kk < 15) a_n = load_fragA(qp + (kk + 1) * 32);
        sacc = wmma_bf16(a, kfr[kk], sacc);
        a = a_n;
      }
      Pack8 pk;
      #pragma unroll
      for (int r = 0; r < 8; r++) {
        int srow = ss0 + r + 8 * hi;
        float mm = Ms[(size_t)b * kHW + srow];
        float ll = Ls[(size_t)b * kHW + srow];
        pk.h[r] = (__bf16)(__expf(sacc[r] - mm) / ll);
      }
      // 8 contiguous s-columns per lane -> single ds_store_b128
      *(v4u*)&pT[lm][w * 32 + half * 16 + 8 * hi] = pk.q;
    }
    __syncthreads();
    #pragma unroll
    for (int i = 0; i < 4; i++) {
      int c0 = w * 64 + 16 * i;
      const __bf16* vp = vb + (size_t)(c0 + lm) * kHW + sbase + kbA;
      v16bf a = load_fragA(vp);
      #pragma unroll
      for (int sub = 0; sub < 8; sub++) {
        v16bf a_n = a;
        if (sub < 7) a_n = load_fragA(vp + (sub + 1) * 32);
        v16bf bf = load_fragB(&pT[lm][sub * 32 + kbB]);
        acc[i] = wmma_bf16(a, bf, acc[i]);
        a = a_n;
      }
    }
    __syncthreads();
  }
  #pragma unroll
  for (int i = 0; i < 4; i++) {
    Pack8 pk;
    #pragma unroll
    for (int r = 0; r < 8; r++) pk.h[r] = (__bf16)acc[i][r];
    // channels c0+8*hi..+7 contiguous in (hw,c) layout -> one b128 store
    *(v4u*)(outT + ((size_t)b * kHW + t0 + lm) * kC + w * 64 + 16 * i + 8 * hi) = pk.q;
  }
}

// ------------------------------------------- proj 1x1 conv + bias + residual
__global__ void __launch_bounds__(256, 1)
proj_residual(const __bf16* __restrict__ Wb,
              const __bf16* __restrict__ outT,
              const float* __restrict__ bias,
              const float* __restrict__ x,
              float* __restrict__ y) {
  int b    = blockIdx.z;
  int lane = threadIdx.x & 31;
  int w    = threadIdx.x >> 5;
  int m0   = blockIdx.y * 128 + w * 16;
  int nb   = blockIdx.x * 64;
  int lm   = lane & 15, hi = lane >> 4;
  int kbA  = hi * 16, kbB = hi * 16;
  const __bf16* Ap  = Wb + (size_t)(m0 + lm) * kC + kbA;
  const __bf16* Btb = outT + (size_t)b * kHW * kC;
  const __bf16* Bp[4];
  #pragma unroll
  for (int i = 0; i < 4; i++)
    Bp[i] = Btb + (size_t)(nb + 16 * i + lm) * kC + kbB;
  v8f acc[4];
  #pragma unroll
  for (int i = 0; i < 4; i++) zero_acc(acc[i]);
  v16bf a = load_fragA(Ap);
  v16bf bfr[4];
  #pragma unroll
  for (int i = 0; i < 4; i++) bfr[i] = load_fragB(Bp[i]);
  #pragma unroll
  for (int kk = 0; kk < 16; kk++) {
    v16bf a_n = a;
    v16bf b_n[4];
    #pragma unroll
    for (int i = 0; i < 4; i++) b_n[i] = bfr[i];
    if (kk < 15) {
      a_n = load_fragA(Ap + (kk + 1) * 32);
      #pragma unroll
      for (int i = 0; i < 4; i++) b_n[i] = load_fragB(Bp[i] + (kk + 1) * 32);
    }
    #pragma unroll
    for (int i = 0; i < 4; i++) acc[i] = wmma_bf16(a, bfr[i], acc[i]);
    a = a_n;
    #pragma unroll
    for (int i = 0; i < 4; i++) bfr[i] = b_n[i];
  }
  #pragma unroll
  for (int i = 0; i < 4; i++) {
    int n0 = nb + 16 * i;
    #pragma unroll
    for (int r = 0; r < 8; r++) {
      int row = m0 + r + 8 * hi;
      size_t idx = ((size_t)b * kC + row) * kHW + n0 + lm;
      y[idx] = acc[i][r] + bias[row] + x[idx];
    }
  }
}

extern "C" void kernel_launch(void* const* d_in, const int* in_sizes, int n_in,
                              void* d_out, int out_size, void* d_ws, size_t ws_size,
                              hipStream_t stream) {
  (void)in_sizes; (void)n_in; (void)out_size; (void)ws_size;
  const float* x     = (const float*)d_in[0];
  const float* gamma = (const float*)d_in[1];
  const float* beta  = (const float*)d_in[2];
  const float* Wq    = (const float*)d_in[3];
  const float* bq    = (const float*)d_in[4];
  const float* Wk    = (const float*)d_in[5];
  const float* bk    = (const float*)d_in[6];
  const float* Wv    = (const float*)d_in[7];
  const float* bv    = (const float*)d_in[8];
  const float* Wp    = (const float*)d_in[9];
  const float* bp    = (const float*)d_in[10];
  float* y = (float*)d_out;

  char* ws = (char*)d_ws;
  size_t off = 0;
  auto alloc = [&](size_t bytes) -> void* {
    void* p = ws + off;
    off += (bytes + 255) & ~(size_t)255;
    return p;
  };
  const size_t wBytes   = (size_t)kC * kC * sizeof(__bf16);
  const size_t actBytes = (size_t)kB * kHW * kC * sizeof(__bf16);
  __bf16* Wqb  = (__bf16*)alloc(wBytes);
  __bf16* Wkb  = (__bf16*)alloc(wBytes);
  __bf16* Wvb  = (__bf16*)alloc(wBytes);
  __bf16* Wpb  = (__bf16*)alloc(wBytes);
  __bf16* Rt   = (__bf16*)alloc(actBytes);   // groupnorm output, (b,hw,c)
  __bf16* qTp  = (__bf16*)alloc(actBytes);   // q^T pre-scaled,   (b,hw,c)
  __bf16* kTp  = (__bf16*)alloc(actBytes);   // k^T,              (b,hw,c)
  __bf16* Vp   = (__bf16*)alloc(actBytes);   // v,                (b,c,hw)
  __bf16* oT   = (__bf16*)alloc(actBytes);   // attn out^T,       (b,hw,c)
  float*  Ms   = (float*)alloc((size_t)kB * kHW * sizeof(float));
  float*  Ls   = (float*)alloc((size_t)kB * kHW * sizeof(float));

  const int nW = kC * kC;
  cvt_f32_bf16<<<dim3(nW / 256), dim3(256), 0, stream>>>(Wq, Wqb, nW);
  cvt_f32_bf16<<<dim3(nW / 256), dim3(256), 0, stream>>>(Wk, Wkb, nW);
  cvt_f32_bf16<<<dim3(nW / 256), dim3(256), 0, stream>>>(Wv, Wvb, nW);
  cvt_f32_bf16<<<dim3(nW / 256), dim3(256), 0, stream>>>(Wp, Wpb, nW);

  gn_kernel<<<dim3(kB * 32), dim3(256), 0, stream>>>(x, gamma, beta, Rt);

  dim3 gGemm(kHW / 64, kC / 128, kB);
  const float qscale = 0.044194173824159216f;  // 1/sqrt(512)
  conv1x1_gemm<<<gGemm, dim3(256), 0, stream>>>(Wqb, Rt, bq, qTp, qscale, 1);
  conv1x1_gemm<<<gGemm, dim3(256), 0, stream>>>(Wkb, Rt, bk, kTp, 1.0f,   1);
  conv1x1_gemm<<<gGemm, dim3(256), 0, stream>>>(Wvb, Rt, bv, Vp,  1.0f,   0);

  attn_stats<<<dim3(kHW / 16, kB), dim3(256), 0, stream>>>(qTp, kTp, Ms, Ls);
  attn_out<<<dim3(kHW / 16, kB), dim3(256), 0, stream>>>(qTp, kTp, Vp, Ms, Ls, oT);

  proj_residual<<<gGemm, dim3(256), 0, stream>>>(Wpb, oT, bp, x, y);
}
